// FlashMHA_24300924961253
// MI455X (gfx1250) — matrix operand order repaired and checked by reading, not compiled, not run on hardware
//
#include <hip/hip_runtime.h>
#include <hip/hip_bf16.h>

typedef __attribute__((ext_vector_type(16))) __bf16 v16bf;
typedef __attribute__((ext_vector_type(8)))  float  v8f;
typedef __attribute__((ext_vector_type(4)))  int    v4i;

#define DIMD 1024
#define SEQL 2048
#define NB   4
#define NH   16
#define HD   64
#define LOG2E 1.44269504088896340736f

union Frag { v16bf v; float4 q[2]; };
union Acc  { v8f v; float f[8]; };

__device__ __forceinline__ unsigned short f2bf(float f) {
  union { float f; unsigned u; } c; c.f = f;
  unsigned r = c.u + 0x7FFFu + ((c.u >> 16) & 1u);
  return (unsigned short)(r >> 16);
}

__device__ __forceinline__ v8f wmma_bf16(v16bf a, v16bf b, v8f c) {
  // D = A(16x32 bf16) x B(32x16 bf16) + C(16x16 f32)
  return __builtin_amdgcn_wmma_f32_16x16x32_bf16(false, a, false, b, (short)0, c,
                                                 false, false);
}

// ------------------------- async global -> LDS copy (CDNA5), 16B per lane ----
typedef __attribute__((address_space(1))) v4i gv4i;   // global int4
typedef __attribute__((address_space(3))) v4i lv4i;   // LDS int4
typedef __attribute__((address_space(3))) void lvoid;

__device__ __forceinline__ void async_copy_b128(const unsigned short* g,
                                                unsigned short* l) {
#if __has_builtin(__builtin_amdgcn_global_load_async_to_lds_b128)
  __builtin_amdgcn_global_load_async_to_lds_b128((gv4i*)(void*)g,
                                                 (lv4i*)(void*)l, 0, 0);
#else
  unsigned loff = (unsigned)(uintptr_t)(lvoid*)l;
  asm volatile("global_load_async_to_lds_b128 %0, %1, off"
               :: "v"(loff), "v"(g) : "memory");
#endif
}

__device__ __forceinline__ void wait_async0() {
#if __has_builtin(__builtin_amdgcn_s_wait_asynccnt)
  __builtin_amdgcn_s_wait_asynccnt(0);
#else
  asm volatile("s_wait_asynccnt 0x0" ::: "memory");
#endif
}

__device__ __forceinline__ float redmax16(float v) {
#pragma unroll
  for (int m = 8; m >= 1; m >>= 1) v = fmaxf(v, __shfl_xor(v, m, 16));
  return v;
}
__device__ __forceinline__ float redsum16(float v) {
#pragma unroll
  for (int m = 8; m >= 1; m >>= 1) v += __shfl_xor(v, m, 16);
  return v;
}

// ---------------------------------------------------------------- fp32 -> bf16
__global__ __launch_bounds__(256) void cvt_bf16(const float* __restrict__ s,
                                                unsigned short* __restrict__ d,
                                                int n4) {
  int i = blockIdx.x * blockDim.x + threadIdx.x;
  if (i >= n4) return;
  float4 f = ((const float4*)s)[i];
  uint2 o;
  o.x = (unsigned)f2bf(f.x) | ((unsigned)f2bf(f.y) << 16);
  o.y = (unsigned)f2bf(f.z) | ((unsigned)f2bf(f.w) << 16);
  ((uint2*)d)[i] = o;
}

// ------------------------------------------------- QKV projection (z selects)
// y[m,n] = sum_k x[m,k] * W[n,k] + bias[n]   (torch Linear)
// Q,K stored [b,h,l,hd] bf16 ; V stored transposed [b,h,hd,l] bf16
__global__ __launch_bounds__(128) void qkv_gemm(
    const unsigned short* __restrict__ xb,
    const unsigned short* __restrict__ wqb,
    const unsigned short* __restrict__ wkb,
    const unsigned short* __restrict__ wvb,
    const float* __restrict__ bq, const float* __restrict__ bk,
    const float* __restrict__ bv,
    unsigned short* __restrict__ Q, unsigned short* __restrict__ K,
    unsigned short* __restrict__ Vt) {
  const int z = blockIdx.z;
  const unsigned short* W = (z == 0) ? wqb : (z == 1) ? wkb : wvb;
  const float* bias = (z == 0) ? bq : (z == 1) ? bk : bv;

  __shared__ unsigned short As[2][64][32];   // double-buffered x tile

  const int tid = threadIdx.x;
  const int w = tid >> 5, lane = tid & 31;
  const int half = lane >> 4, nl = lane & 15;
  const int row0 = blockIdx.x * 64;
  const int col0 = blockIdx.y * 128 + w * 32;

  Acc acc[4][2];
#pragma unroll
  for (int mt = 0; mt < 4; ++mt)
#pragma unroll
    for (int nt = 0; nt < 2; ++nt)
#pragma unroll
      for (int r = 0; r < 8; ++r) acc[mt][nt].f[r] = 0.f;

  // async stage of a 64x32 bf16 x-tile: 256 16B chunks, 2 per thread
  const int cr0 = (tid + 0) >> 2,   cc0 = ((tid + 0) & 3) * 8;
  const int cr1 = (tid + 128) >> 2, cc1 = ((tid + 128) & 3) * 8;
#define COPY_X(buf, k0)                                                        \
  do {                                                                         \
    async_copy_b128(xb + (size_t)(row0 + cr0) * DIMD + (k0) + cc0,             \
                    &As[buf][cr0][cc0]);                                       \
    async_copy_b128(xb + (size_t)(row0 + cr1) * DIMD + (k0) + cc1,             \
                    &As[buf][cr1][cc1]);                                       \
  } while (0)

  COPY_X(0, 0);
  for (int k0 = 0; k0 < DIMD; k0 += 32) {
    const int buf = (k0 >> 5) & 1;
    wait_async0();
    __syncthreads();                       // tile `buf` visible to all waves
    if (k0 + 32 < DIMD) COPY_X(buf ^ 1, k0 + 32);

    Frag bf[2];
#pragma unroll
    for (int nt = 0; nt < 2; ++nt) {       // B frag: 16 contiguous K per lane
      const unsigned short* p =
          W + (size_t)(col0 + nt * 16 + nl) * DIMD + k0 + half * 16;
      bf[nt].q[0] = *(const float4*)p;
      bf[nt].q[1] = *(const float4*)(p + 8);
    }
#pragma unroll
    for (int mt = 0; mt < 4; ++mt) {       // A frag from LDS (16 contiguous K per half)
      Frag af;
      const unsigned short* p = &As[buf][mt * 16 + nl][half * 16];
      af.q[0] = *(const float4*)p;
      af.q[1] = *(const float4*)(p + 8);
#pragma unroll
      for (int nt = 0; nt < 2; ++nt)
        acc[mt][nt].v = wmma_bf16(af.v, bf[nt].v, acc[mt][nt].v);
    }
  }
#undef COPY_X

#pragma unroll
  for (int nt = 0; nt < 2; ++nt) {
    const int n = col0 + nt * 16 + nl;
    const float bb = bias[n];
    const int h = n >> 6, hd = n & 63;
#pragma unroll
    for (int mt = 0; mt < 4; ++mt) {
      if (z == 2) {                        // V: transposed, dword-packed stores
#pragma unroll
        for (int r = 0; r < 8; r += 2) {
          const int m = row0 + mt * 16 + half * 8 + r;
          const int b = m >> 11, l = m & (SEQL - 1);
          unsigned pk = (unsigned)f2bf(acc[mt][nt].f[r] + bb) |
                        ((unsigned)f2bf(acc[mt][nt].f[r + 1] + bb) << 16);
          *(unsigned*)&Vt[((size_t)(b * NH + h) * HD + hd) * SEQL + l] = pk;
        }
      } else {
#pragma unroll
        for (int r = 0; r < 8; ++r) {
          const int m = row0 + mt * 16 + half * 8 + r;
          const int b = m >> 11, l = m & (SEQL - 1);
          ((z == 0) ? Q : K)[((size_t)(b * NH + h) * SEQL + l) * HD + hd] =
              f2bf(acc[mt][nt].f[r] + bb);
        }
      }
    }
  }
}

// --------------------------------------------- causal flash attention + ALiBi
__global__ __launch_bounds__(128) void attn_kernel(
    const unsigned short* __restrict__ Q, const unsigned short* __restrict__ K,
    const unsigned short* __restrict__ Vt, unsigned short* __restrict__ O) {
  __shared__ unsigned short Kt[2][64 * 64];   // K tile  [token][hd]
  __shared__ unsigned short Vtl[2][64 * 64];  // V tile  [hd][token]
  __shared__ unsigned short Pl[4][16 * 64];   // wave-private P slabs

  const int tid = threadIdx.x;
  const int w = tid >> 5, lane = tid & 31;
  const int half = lane >> 4, nl = lane & 15;
  const int qblk = blockIdx.x;
  const int bh = blockIdx.y;
  const int b = bh >> 4, h = bh & 15;

  const unsigned short* Qb = Q + (size_t)bh * SEQL * HD;
  const unsigned short* Kb = K + (size_t)bh * SEQL * HD;
  const unsigned short* Vb = Vt + (size_t)bh * HD * SEQL;

  const int q0 = qblk * 64 + w * 16;

  Frag qf[2];                                // Q rows as A frags (hd = 2x K=32)
#pragma unroll
  for (int ks = 0; ks < 2; ++ks) {
    const unsigned short* p = Qb + (size_t)(q0 + nl) * HD + ks * 32 + half * 16;
    qf[ks].q[0] = *(const float4*)p;
    qf[ks].q[1] = *(const float4*)(p + 8);
  }

  const float slope2 = exp2f(-0.5f * (float)(h + 1)) * LOG2E;  // ALiBi (log2)
  const float sc2 = 0.125f * LOG2E;                            // 1/sqrt(64)

  Acc o[4];
  float rowM[8], rowL[8];
#pragma unroll
  for (int r = 0; r < 8; ++r) { rowM[r] = -1e30f; rowL[r] = 0.f; }
#pragma unroll
  for (int nt = 0; nt < 4; ++nt)
#pragma unroll
    for (int r = 0; r < 8; ++r) o[nt].f[r] = 0.f;

  // async stage of 64x64 K and V tiles: 512 16B chunks each, 4/thread each
#define COPY_KV(buf, kbase)                                                    \
  do {                                                                         \
    _Pragma("unroll") for (int i = 0; i < 4; ++i) {                            \
      const int c = tid + i * 128;                                             \
      const int r = c >> 3, col = (c & 7) * 8;                                 \
      async_copy_b128(Kb + (size_t)((kbase) + r) * HD + col,                   \
                      &Kt[buf][r * 64 + col]);                                 \
      async_copy_b128(Vb + (size_t)r * SEQL + (kbase) + col,                   \
                      &Vtl[buf][r * 64 + col]);                                \
    }                                                                          \
  } while (0)

  COPY_KV(0, 0);
  for (int kb = 0; kb <= qblk; ++kb) {       // causal: only lower blocks
    const int buf = kb & 1;
    const int kbase = kb * 64;
    wait_async0();
    __syncthreads();                         // K/V tile `buf` ready
    if (kb < qblk) COPY_KV(buf ^ 1, kbase + 64);

    Acc s[4];                                // S = Q K^T, 16x64 tile
#pragma unroll
    for (int nt = 0; nt < 4; ++nt) {
#pragma unroll
      for (int r = 0; r < 8; ++r) s[nt].f[r] = 0.f;
#pragma unroll
      for (int ks = 0; ks < 2; ++ks) {
        Frag kf;                             // K rows from LDS as B frags
        const unsigned short* p =
            &Kt[buf][(nt * 16 + nl) * 64 + ks * 32 + half * 16];
        kf.q[0] = *(const float4*)p;
        kf.q[1] = *(const float4*)(p + 8);
        s[nt].v = wmma_bf16(qf[ks].v, kf.v, s[nt].v);
      }
    }

    float bm[8];
#pragma unroll
    for (int r = 0; r < 8; ++r) bm[r] = -1e30f;
#pragma unroll
    for (int nt = 0; nt < 4; ++nt) {
      const int j = kbase + nt * 16 + nl;
#pragma unroll
      for (int r = 0; r < 8; ++r) {
        const int i = q0 + half * 8 + r;
        float v = s[nt].f[r] * sc2 - slope2 * (float)(i - j);
        if (j > i) v = -1e30f;               // causal mask (diag block only)
        s[nt].f[r] = v;
        bm[r] = fmaxf(bm[r], v);
      }
    }
#pragma unroll
    for (int r = 0; r < 8; ++r) bm[r] = redmax16(bm[r]);

    float corr[8];
#pragma unroll
    for (int r = 0; r < 8; ++r) {
      float mn = fmaxf(rowM[r], bm[r]);
      corr[r] = exp2f(rowM[r] - mn);
      rowM[r] = mn;
      rowL[r] *= corr[r];
    }
#pragma unroll
    for (int nt = 0; nt < 4; ++nt)
#pragma unroll
      for (int r = 0; r < 8; ++r) o[nt].f[r] *= corr[r];

    float ps[8];
#pragma unroll
    for (int r = 0; r < 8; ++r) ps[r] = 0.f;
#pragma unroll
    for (int nt = 0; nt < 4; ++nt)
#pragma unroll
      for (int r = 0; r < 8; ++r) {
        float p = exp2f(s[nt].f[r] - rowM[r]);
        ps[r] += p;
        // C-layout -> LDS (bf16), so we can reload in A-layout
        Pl[w][(half * 8 + r) * 64 + nt * 16 + nl] = f2bf(p);
      }
#pragma unroll
    for (int r = 0; r < 8; ++r) rowL[r] += redsum16(ps[r]);

#pragma unroll
    for (int ks = 0; ks < 2; ++ks) {         // O += P * V
      Frag pf;                               // A frag of P from LDS
      const unsigned short* pp = &Pl[w][nl * 64 + ks * 32 + half * 16];
      pf.q[0] = *(const float4*)pp;
      pf.q[1] = *(const float4*)(pp + 8);
#pragma unroll
      for (int nt = 0; nt < 4; ++nt) {
        Frag vf;                             // V^T rows: 16 contiguous tokens
        const unsigned short* vp =
            &Vtl[buf][(nt * 16 + nl) * 64 + ks * 32 + half * 16];
        vf.q[0] = *(const float4*)vp;
        vf.q[1] = *(const float4*)(vp + 8);
        o[nt].v = wmma_bf16(pf.v, vf.v, o[nt].v);
      }
    }
  }
#undef COPY_KV

  float inv[8];
#pragma unroll
  for (int r = 0; r < 8; ++r) inv[r] = 1.0f / rowL[r];
#pragma unroll
  for (int nt = 0; nt < 4; ++nt)
#pragma unroll
    for (int r = 0; r < 8; ++r) {
      const int i = q0 + half * 8 + r;
      O[(size_t)(b * SEQL + i) * DIMD + h * HD + nt * 16 + nl] =
          f2bf(o[nt].f[r] * inv[r]);
    }
}

// --------------------------------------------------- output projection (fp32)
__global__ __launch_bounds__(128) void out_proj(
    const unsigned short* __restrict__ Ob, const unsigned short* __restrict__ wob,
    const float* __restrict__ bo, float* __restrict__ out) {
  __shared__ unsigned short As[2][64][32];

  const int tid = threadIdx.x;
  const int w = tid >> 5, lane = tid & 31;
  const int half = lane >> 4, nl = lane & 15;
  const int row0 = blockIdx.x * 64;
  const int col0 = blockIdx.y * 128 + w * 32;

  Acc acc[4][2];
#pragma unroll
  for (int mt = 0; mt < 4; ++mt)
#pragma unroll
    for (int nt = 0; nt < 2; ++nt)
#pragma unroll
      for (int r = 0; r < 8; ++r) acc[mt][nt].f[r] = 0.f;

  const int cr0 = (tid + 0) >> 2,   cc0 = ((tid + 0) & 3) * 8;
  const int cr1 = (tid + 128) >> 2, cc1 = ((tid + 128) & 3) * 8;
#define COPY_O(buf, k0)                                                        \
  do {                                                                         \
    async_copy_b128(Ob + (size_t)(row0 + cr0) * DIMD + (k0) + cc0,             \
                    &As[buf][cr0][cc0]);                                       \
    async_copy_b128(Ob + (size_t)(row0 + cr1) * DIMD + (k0) + cc1,             \
                    &As[buf][cr1][cc1]);                                       \
  } while (0)

  COPY_O(0, 0);
  for (int k0 = 0; k0 < DIMD; k0 += 32) {
    const int buf = (k0 >> 5) & 1;
    wait_async0();
    __syncthreads();
    if (k0 + 32 < DIMD) COPY_O(buf ^ 1, k0 + 32);

    Frag bf[2];
#pragma unroll
    for (int nt = 0; nt < 2; ++nt) {
      const unsigned short* p =
          wob + (size_t)(col0 + nt * 16 + nl) * DIMD + k0 + half * 16;
      bf[nt].q[0] = *(const float4*)p;
      bf[nt].q[1] = *(const float4*)(p + 8);
    }
#pragma unroll
    for (int mt = 0; mt < 4; ++mt) {
      Frag af;
      const unsigned short* p = &As[buf][mt * 16 + nl][half * 16];
      af.q[0] = *(const float4*)p;
      af.q[1] = *(const float4*)(p + 8);
#pragma unroll
      for (int nt = 0; nt < 2; ++nt)
        acc[mt][nt].v = wmma_bf16(af.v, bf[nt].v, acc[mt][nt].v);
    }
  }
#undef COPY_O

#pragma unroll
  for (int nt = 0; nt < 2; ++nt) {
    const int n = col0 + nt * 16 + nl;
    const float bb = bo[n];
#pragma unroll
    for (int mt = 0; mt < 4; ++mt)
#pragma unroll
      for (int r = 0; r < 8; ++r) {
        const int m = row0 + mt * 16 + half * 8 + r;
        out[(size_t)m * DIMD + n] = acc[mt][nt].f[r] + bb;
      }
  }
}

// ----------------------------------------------------------------- dispatcher
extern "C" void kernel_launch(void* const* d_in, const int* in_sizes, int n_in,
                              void* d_out, int out_size, void* d_ws,
                              size_t ws_size, hipStream_t stream) {
  const float* x  = (const float*)d_in[0];
  const float* Wq = (const float*)d_in[1];
  const float* bq = (const float*)d_in[2];
  const float* Wk = (const float*)d_in[3];
  const float* bk = (const float*)d_in[4];
  const float* Wv = (const float*)d_in[5];
  const float* bv = (const float*)d_in[6];
  const float* Wo = (const float*)d_in[7];
  const float* bo = (const float*)d_in[8];

  unsigned short* ws = (unsigned short*)d_ws;
  // workspace layout (bf16 element offsets)
  unsigned short* XB  = ws + 0;                // 8,388,608  x
  unsigned short* WQ  = ws + 8388608;          // 1,048,576
  unsigned short* WK  = ws + 9437184;
  unsigned short* WV  = ws + 10485760;
  unsigned short* WOB = ws + 11534336;
  unsigned short* Qb  = ws + 12582912;         // [b,h,l,hd]
  unsigned short* Kb  = ws + 20971520;         // [b,h,l,hd]
  unsigned short* Vt  = ws + 29360128;         // [b,h,hd,l]
  unsigned short* Ob  = ws + 37748736;         // [b,l,d]

  const int nx4 = (NB * SEQL * DIMD) / 4;      // 2,097,152
  const int nw4 = (DIMD * DIMD) / 4;           //   262,144
  cvt_bf16<<<(nx4 + 255) / 256, 256, 0, stream>>>(x, XB, nx4);
  cvt_bf16<<<(nw4 + 255) / 256, 256, 0, stream>>>(Wq, WQ, nw4);
  cvt_bf16<<<(nw4 + 255) / 256, 256, 0, stream>>>(Wk, WK, nw4);
  cvt_bf16<<<(nw4 + 255) / 256, 256, 0, stream>>>(Wv, WV, nw4);
  cvt_bf16<<<(nw4 + 255) / 256, 256, 0, stream>>>(Wo, WOB, nw4);

  qkv_gemm<<<dim3((NB * SEQL) / 64, DIMD / 128, 3), 128, 0, stream>>>(
      XB, WQ, WK, WV, bq, bk, bv, Qb, Kb, Vt);

  attn_kernel<<<dim3(SEQL / 64, NB * NH), 128, 0, stream>>>(Qb, Kb, Vt, Ob);

  out_proj<<<dim3((NB * SEQL) / 64, DIMD / 128), 128, 0, stream>>>(
      Ob, WOB, bo, (float*)d_out);
}
